// MultiHeadAttention_10075993276483
// MI455X (gfx1250) — matrix operand order repaired and checked by reading, not compiled, not run on hardware
//
#include <hip/hip_runtime.h>
#include <hip/hip_bf16.h>
#include <math.h>

// MI455X / gfx1250, wave32. bf16 WMMA path: v_wmma_f32_16x16x32_bf16.
typedef __bf16 bf16;
typedef __attribute__((ext_vector_type(16))) __bf16 v16bf;
typedef __attribute__((ext_vector_type(8)))  float  v8f;

#define BB 4
#define TT 2048
#define CC 768
#define HH 12
#define HSZ 64

// ---- fragment loaders, contiguous k = e + 16*half for both A and B ----------------------

// 16-bit A-matrix 16x32 (MxK). Lane L: M = L%16, half = L/16.
// elements 0..15 -> K = e + 16*half
__device__ __forceinline__ v16bf load_a16x32(const bf16* __restrict__ base, int ld) {
  const int lane = threadIdx.x & 31;
  const int m    = lane & 15;
  const int half = lane >> 4;
  const bf16* row = base + (size_t)m * ld;
  v16bf a;
#pragma unroll
  for (int e = 0; e < 8; ++e)  a[e] = row[e + 16 * half];
#pragma unroll
  for (int e = 8; e < 16; ++e) a[e] = row[e + 16 * half];
  return a;
}

// 16-bit B-matrix 32x16 (KxN). Lane L: N = L%16, elements e -> K = e + 16*(L/16).
// Source: row n contiguous over k (i.e. B(k,n) = base[n*ld + k]).
__device__ __forceinline__ v16bf load_b32x16(const bf16* __restrict__ base, int ld) {
  const int lane = threadIdx.x & 31;
  const int n    = lane & 15;
  const int kb   = (lane >> 4) * 16;
  const bf16* row = base + (size_t)n * ld + kb;
  v16bf b;
#pragma unroll
  for (int e = 0; e < 16; ++e) b[e] = row[e];
  return b;
}

__device__ __forceinline__ v8f wmma_bf16(v16bf a, v16bf b, v8f c) {
  return __builtin_amdgcn_wmma_f32_16x16x32_bf16(false, a, false, b, (short)0, c,
                                                 false, false);
}

// ---- packing / conversion kernels ---------------------------------------

__global__ void cvt_f32_bf16(const float* __restrict__ src, bf16* __restrict__ dst,
                             int n) {
  int i = blockIdx.x * blockDim.x + threadIdx.x;
  if (i < n) dst[i] = (bf16)src[i];
}

// Wq/Wk/Wv [H, C, HS] fp32 -> Wcat bf16 [3*C, C], row n = sel*C + h*HS + d,
// contiguous over c (so B-fragment loads are contiguous).
__global__ void pack_qkv_w(const float* __restrict__ Wq, const float* __restrict__ Wk,
                           const float* __restrict__ Wv, bf16* __restrict__ Wcat) {
  int i = blockIdx.x * blockDim.x + threadIdx.x;
  const int total = 3 * CC * CC;
  if (i >= total) return;
  int n = i / CC;
  int c = i - n * CC;
  int sel = n / CC;
  int rem = n - sel * CC;
  int h = rem / HSZ;
  int d = rem - h * HSZ;
  const float* W = (sel == 0) ? Wq : (sel == 1) ? Wk : Wv;
  Wcat[(size_t)n * CC + c] = (bf16)W[((size_t)h * CC + c) * HSZ + d];
}

// ---- fused QKV projection GEMM: [8192 x 768] x [768 x 2304] -------------
// 32x64 tile per wave: 2 A-frags x 4 B-frags -> 8 WMMAs per 32-deep K step.
// All 4 waves of a block share the M-strip (A hits L0). Q gets 1/sqrt(HS)
// folded in; V stored transposed [HS, T] so attention PV B-loads are dense.
__global__ void qkv_gemm(const bf16* __restrict__ xb, const bf16* __restrict__ Wcat,
                         bf16* __restrict__ Qp, bf16* __restrict__ Kp,
                         bf16* __restrict__ Vt) {
  const int wave = threadIdx.x >> 5;
  const int lane = threadIdx.x & 31;
  const int half = lane >> 4;
  const int nl   = lane & 15;
  const int tile_m = blockIdx.x * 32;
  const int tile_n = (blockIdx.y * 4 + wave) * 64;   // 64-aligned => one (sel,h)
  v8f acc[2][4] = {};
  const bf16* arow = xb + (size_t)tile_m * CC;
  const bf16* brow = Wcat + (size_t)tile_n * CC;
#pragma unroll 2
  for (int kk = 0; kk < CC; kk += 32) {
    v16bf a0 = load_a16x32(arow + kk, CC);
    v16bf a1 = load_a16x32(arow + (size_t)16 * CC + kk, CC);
#pragma unroll
    for (int j = 0; j < 4; ++j) {
      v16bf b = load_b32x16(brow + (size_t)(16 * j) * CC + kk, CC);
      acc[0][j] = wmma_bf16(a0, b, acc[0][j]);
      acc[1][j] = wmma_bf16(a1, b, acc[1][j]);
    }
  }
  const int sel = tile_n / CC;
  const int rem = tile_n - sel * CC;
  const int h = rem / HSZ;           // rem % 64 == 0: wave covers a full head dim
#pragma unroll
  for (int i = 0; i < 2; ++i) {
#pragma unroll
    for (int r = 0; r < 8; ++r) {
      int m  = tile_m + 16 * i + r + 8 * half;  // C/D layout: M = r + 8*(lane/16)
      int b_ = m / TT, t = m - b_ * TT;
      int bh = b_ * HH + h;
#pragma unroll
      for (int j = 0; j < 4; ++j) {
        int d = 16 * j + nl;
        float v = acc[i][j][r];
        if (sel == 0)      Qp[((size_t)bh * TT + t) * HSZ + d] = (bf16)(v * 0.125f);
        else if (sel == 1) Kp[((size_t)bh * TT + t) * HSZ + d] = (bf16)v;
        else               Vt[((size_t)bh * HSZ + d) * TT + t] = (bf16)v;
      }
    }
  }
}

// ---- causal flash attention ---------------------------------------------
// Block = 128 threads = 4 waves; each wave owns a 16-row Q strip and streams
// 64-wide KV blocks with online softmax: 8 score WMMAs + 8 PV WMMAs per step.
__global__ void attn_kernel(const bf16* __restrict__ Qp, const bf16* __restrict__ Kp,
                            const bf16* __restrict__ Vt, bf16* __restrict__ Ob) {
  __shared__ bf16 lds[4][16][72];   // stride 72 bf16 = 144 B (odd bank stride)
  const int wave = threadIdx.x >> 5;
  const int lane = threadIdx.x & 31;
  const int half = lane >> 4;
  const int nl   = lane & 15;
  const int bh = blockIdx.y;
  const int q0 = blockIdx.x * 64 + wave * 16;
  const bf16* Qb = Qp + (size_t)bh * TT * HSZ;
  const bf16* Kb = Kp + (size_t)bh * TT * HSZ;
  const bf16* Vb = Vt + (size_t)bh * HSZ * TT;

  v16bf qa0 = load_a16x32(Qb + (size_t)q0 * HSZ + 0, HSZ);
  v16bf qa1 = load_a16x32(Qb + (size_t)q0 * HSZ + 32, HSZ);

  v8f o[4] = {};
  float mi[8], li[8];
#pragma unroll
  for (int r = 0; r < 8; ++r) { mi[r] = -__builtin_inff(); li[r] = 0.f; }

  const int nblk = (q0 >> 6) + 1;   // causal bound; diag block never fully masked
  for (int sb = 0; sb < nblk; ++sb) {
    const int s0 = sb * 64;
    if (sb + 1 < nblk) {            // global_prefetch_b8 for next K block
      __builtin_prefetch(Kb + (size_t)(s0 + 64) * HSZ + lane * 8, 0, 0);
      __builtin_prefetch(Kb + (size_t)(s0 + 96) * HSZ + lane * 8, 0, 0);
    }

    // S = Q K^T : four 16x16 tiles, each K-reduced over d=64 (2 WMMAs)
    v8f st[4];
#pragma unroll
    for (int j = 0; j < 4; ++j) {
      v16bf b0 = load_b32x16(Kb + (size_t)(s0 + 16 * j) * HSZ + 0, HSZ);
      v16bf b1 = load_b32x16(Kb + (size_t)(s0 + 16 * j) * HSZ + 32, HSZ);
      v8f t = {};
      t = wmma_bf16(qa0, b0, t);
      t = wmma_bf16(qa1, b1, t);
      st[j] = t;
    }

    // online softmax per row; restage exp(S) into LDS in A layout order
#pragma unroll
    for (int r = 0; r < 8; ++r) {
      const int trow = q0 + r + 8 * half;
      float v0 = (s0 +  0 + nl <= trow) ? st[0][r] : -__builtin_inff();
      float v1 = (s0 + 16 + nl <= trow) ? st[1][r] : -__builtin_inff();
      float v2 = (s0 + 32 + nl <= trow) ? st[2][r] : -__builtin_inff();
      float v3 = (s0 + 48 + nl <= trow) ? st[3][r] : -__builtin_inff();
      float rm = fmaxf(fmaxf(v0, v1), fmaxf(v2, v3));
#pragma unroll
      for (int off = 8; off; off >>= 1) rm = fmaxf(rm, __shfl_xor(rm, off, 16));
      float mnew = fmaxf(mi[r], rm);
      float a = __expf(mi[r] - mnew);
      float p0 = __expf(v0 - mnew);
      float p1 = __expf(v1 - mnew);
      float p2 = __expf(v2 - mnew);
      float p3 = __expf(v3 - mnew);
      const int row = r + 8 * half;
      lds[wave][row][ 0 + nl] = (bf16)p0;
      lds[wave][row][16 + nl] = (bf16)p1;
      lds[wave][row][32 + nl] = (bf16)p2;
      lds[wave][row][48 + nl] = (bf16)p3;
      float rs = (p0 + p1) + (p2 + p3);
#pragma unroll
      for (int off = 8; off; off >>= 1) rs += __shfl_xor(rs, off, 16);
      li[r] = li[r] * a + rs;
      mi[r] = mnew;
      o[0][r] *= a; o[1][r] *= a; o[2][r] *= a; o[3][r] *= a;
    }

    // P V : two A-frags (s 0..31 / 32..63) x four d-tiles
    // same-wave DS ops are in-order; no barrier needed
    v16bf pa0 = load_a16x32(&lds[wave][0][0],  72);
    v16bf pa1 = load_a16x32(&lds[wave][0][32], 72);
#pragma unroll
    for (int jd = 0; jd < 4; ++jd) {
      v16bf bv0 = load_b32x16(Vb + (size_t)(16 * jd) * TT + s0,      TT);
      v16bf bv1 = load_b32x16(Vb + (size_t)(16 * jd) * TT + s0 + 32, TT);
      o[jd] = wmma_bf16(pa0, bv0, o[jd]);
      o[jd] = wmma_bf16(pa1, bv1, o[jd]);
    }
  }

  // normalize and write head-concatenated [B*T, C] bf16 (transpose is free)
  const int b_ = bh / HH, h = bh - b_ * HH;
#pragma unroll
  for (int r = 0; r < 8; ++r) {
    const int t = q0 + r + 8 * half;
    const float inv = 1.0f / li[r];
    bf16* orow = Ob + ((size_t)(b_ * TT + t)) * CC + h * HSZ;
    orow[ 0 + nl] = (bf16)(o[0][r] * inv);
    orow[16 + nl] = (bf16)(o[1][r] * inv);
    orow[32 + nl] = (bf16)(o[2][r] * inv);
    orow[48 + nl] = (bf16)(o[3][r] * inv);
  }
}

// ---- output projection: out = O @ Wp^T + bp (fp32 out), 32x64 per wave --
__global__ void proj_gemm(const bf16* __restrict__ Ob, const bf16* __restrict__ Wpb,
                          const float* __restrict__ bp, float* __restrict__ out) {
  const int wave = threadIdx.x >> 5;
  const int lane = threadIdx.x & 31;
  const int half = lane >> 4;
  const int nl   = lane & 15;
  const int tile_m = blockIdx.x * 32;
  const int tile_n = (blockIdx.y * 4 + wave) * 64;
  v8f acc[2][4] = {};
  const bf16* arow = Ob + (size_t)tile_m * CC;
  const bf16* brow = Wpb + (size_t)tile_n * CC;  // Wp[n][k]: row n contiguous in k
#pragma unroll 2
  for (int kk = 0; kk < CC; kk += 32) {
    v16bf a0 = load_a16x32(arow + kk, CC);
    v16bf a1 = load_a16x32(arow + (size_t)16 * CC + kk, CC);
#pragma unroll
    for (int j = 0; j < 4; ++j) {
      v16bf b = load_b32x16(brow + (size_t)(16 * j) * CC + kk, CC);
      acc[0][j] = wmma_bf16(a0, b, acc[0][j]);
      acc[1][j] = wmma_bf16(a1, b, acc[1][j]);
    }
  }
#pragma unroll
  for (int j = 0; j < 4; ++j) {
    const int n = tile_n + 16 * j + nl;
    const float bias = bp[n];
#pragma unroll
    for (int i = 0; i < 2; ++i) {
#pragma unroll
      for (int r = 0; r < 8; ++r) {
        int m = tile_m + 16 * i + r + 8 * half;
        out[(size_t)m * CC + n] = acc[i][j][r] + bias;
      }
    }
  }
}

// ---- host launcher -------------------------------------------------------
extern "C" void kernel_launch(void* const* d_in, const int* in_sizes, int n_in,
                              void* d_out, int out_size, void* d_ws, size_t ws_size,
                              hipStream_t stream) {
  (void)in_sizes; (void)n_in; (void)out_size; (void)ws_size;
  const float* x  = (const float*)d_in[0];
  const float* Wq = (const float*)d_in[1];
  const float* Wk = (const float*)d_in[2];
  const float* Wv = (const float*)d_in[3];
  const float* Wp = (const float*)d_in[4];
  const float* bp = (const float*)d_in[5];
  float* out = (float*)d_out;

  char* ws = (char*)d_ws;
  size_t off = 0;
  auto take = [&](size_t bytes) -> void* {
    void* p = ws + off;
    off += (bytes + 255) & ~(size_t)255;
    return p;
  };
  const size_t MT = (size_t)BB * TT;                 // 8192 tokens
  bf16* xb   = (bf16*)take(MT * CC * 2);             // x in bf16
  bf16* Wcat = (bf16*)take((size_t)3 * CC * CC * 2); // QKV weights [2304,768]
  bf16* Wpb  = (bf16*)take((size_t)CC * CC * 2);     // Wp bf16
  bf16* Qp   = (bf16*)take(MT * CC * 2);             // [B,H,T,HS], pre-scaled
  bf16* Kp   = (bf16*)take(MT * CC * 2);             // [B,H,T,HS]
  bf16* Vt   = (bf16*)take(MT * CC * 2);             // [B,H,HS,T] (transposed)
  bf16* Ob   = (bf16*)take(MT * CC * 2);             // attention out [B*T,C]

  int nx = (int)(MT * CC);
  cvt_f32_bf16<<<(nx + 255) / 256, 256, 0, stream>>>(x, xb, nx);
  int nwp = CC * CC;
  cvt_f32_bf16<<<(nwp + 255) / 256, 256, 0, stream>>>(Wp, Wpb, nwp);
  int nwc = 3 * CC * CC;
  pack_qkv_w<<<(nwc + 255) / 256, 256, 0, stream>>>(Wq, Wk, Wv, Wcat);

  // 32x64 per wave, 4 waves/block: grid = (8192/32, 2304/256) and (…, 768/256)
  qkv_gemm<<<dim3((unsigned)(MT / 32), (3 * CC) / 256), 128, 0, stream>>>(
      xb, Wcat, Qp, Kp, Vt);
  attn_kernel<<<dim3(TT / 64, BB * HH), 128, 0, stream>>>(Qp, Kp, Vt, Ob);
  proj_gemm<<<dim3((unsigned)(MT / 32), CC / 256), 128, 0, stream>>>(
      Ob, Wpb, bp, out);
}
